// WindowSpatialAttention_68341519614056
// MI455X (gfx1250) — compile-verified, matrix operand order checked by reading, not run on hardware
//
#include <hip/hip_runtime.h>
#include <hip/hip_bf16.h>

// ---------------------------------------------------------------------------
// CDNA5 (gfx1250) WMMA types
// ---------------------------------------------------------------------------
typedef __attribute__((ext_vector_type(16))) __bf16 v16bf;
typedef __attribute__((ext_vector_type(8)))  __bf16 v8bf;
typedef __attribute__((ext_vector_type(4)))  __bf16 v4bf;
typedef __attribute__((ext_vector_type(8)))  float  v8f;

// Load a 16-bit A/B fragment whose per-lane K values are contiguous in memory:
// two b128 loads + register concat. base must be 16B aligned.
__device__ __forceinline__ v16bf ld_frag_bf(const __bf16* __restrict__ base, int hi) {
    v8bf lo = *(const v8bf*)(base + (hi << 3));
    v8bf hh = *(const v8bf*)(base + 16 + (hi << 3));
    return __builtin_shufflevector(lo, hh, 0, 1, 2, 3, 4, 5, 6, 7,
                                          8, 9, 10, 11, 12, 13, 14, 15);
}

// Same, but source is f32 with a scale factor (acc/cnt path): 4x b128 + cvt_pk.
__device__ __forceinline__ v16bf ld_frag_f32(const float* __restrict__ base, int hi,
                                             float scale) {
    float4 a0 = ((const float4*)(base + (hi << 3)))[0];
    float4 a1 = ((const float4*)(base + (hi << 3)))[1];
    float4 b0 = ((const float4*)(base + 16 + (hi << 3)))[0];
    float4 b1 = ((const float4*)(base + 16 + (hi << 3)))[1];
    v16bf r;
    r[0]  = (__bf16)(a0.x * scale); r[1]  = (__bf16)(a0.y * scale);
    r[2]  = (__bf16)(a0.z * scale); r[3]  = (__bf16)(a0.w * scale);
    r[4]  = (__bf16)(a1.x * scale); r[5]  = (__bf16)(a1.y * scale);
    r[6]  = (__bf16)(a1.z * scale); r[7]  = (__bf16)(a1.w * scale);
    r[8]  = (__bf16)(b0.x * scale); r[9]  = (__bf16)(b0.y * scale);
    r[10] = (__bf16)(b0.z * scale); r[11] = (__bf16)(b0.w * scale);
    r[12] = (__bf16)(b1.x * scale); r[13] = (__bf16)(b1.y * scale);
    r[14] = (__bf16)(b1.z * scale); r[15] = (__bf16)(b1.w * scale);
    return r;
}

// concat two v4bf into v8bf
__device__ __forceinline__ v8bf cat44(v4bf a, v4bf b) {
    return __builtin_shufflevector(a, b, 0, 1, 2, 3, 4, 5, 6, 7);
}

// Problem constants (fixed by the harness)
#define BB   2
#define YY   128
#define XX   128
#define CC   128
#define HH   8
#define DD   32
#define HF   256
#define NWIN 31              // windows per axis, start = 4*i
#define NPIX (BB * YY * XX)  // 32768

// ---------------------------------------------------------------------------
// Prep kernels
// ---------------------------------------------------------------------------
__global__ void zero_kernel(float* __restrict__ p, size_t n) {
    size_t i = (size_t)blockIdx.x * blockDim.x + threadIdx.x;
    size_t stride = (size_t)gridDim.x * blockDim.x;
    for (; i < n; i += stride) p[i] = 0.0f;
}

// f32 -> bf16, 4 elements per thread (b128 in, b64 out)
__global__ void cvt4_kernel(const float* __restrict__ src, __bf16* __restrict__ dst,
                            size_t n4) {
    size_t i = (size_t)blockIdx.x * blockDim.x + threadIdx.x;
    if (i >= n4) return;
    float4 t = ((const float4*)src)[i];
    v4bf o;
    o[0] = (__bf16)t.x; o[1] = (__bf16)t.y; o[2] = (__bf16)t.z; o[3] = (__bf16)t.w;
    ((v4bf*)dst)[i] = o;
}

// W (K x N, f32, row-major) -> Wt (N x K, bf16): makes B fragments contiguous
__global__ void transpose_kernel(const float* __restrict__ W, __bf16* __restrict__ Wt,
                                 int K, int N) {
    int i = blockIdx.x * blockDim.x + threadIdx.x;
    if (i >= K * N) return;
    int n = i / K, k = i % K;               // output-major
    Wt[i] = (__bf16)W[(size_t)k * N + n];
}

// ---------------------------------------------------------------------------
// Kernel 1: fused QKV projection GEMM  (M=32768, K=128, N=256) x3
// register-blocked 16M x 64N per wave: 1 A-frag -> 4 WMMAs per k-step.
// Q,K stored [pixel][feature]; V stored transposed [feature][pixel].
// ---------------------------------------------------------------------------
__global__ __launch_bounds__(256) void qkv_kernel(
    const __bf16* __restrict__ xb,
    const __bf16* __restrict__ WtQ, const float* __restrict__ bq,
    const __bf16* __restrict__ WtK, const float* __restrict__ bk,
    const __bf16* __restrict__ WtV, const float* __restrict__ bv,
    __bf16* __restrict__ qb, __bf16* __restrict__ kb, __bf16* __restrict__ vbT)
{
    int wave = (int)((blockIdx.x * blockDim.x + threadIdx.x) >> 5);
    int lane = threadIdx.x & 31;
    int hi = lane >> 4, lm = lane & 15;

    int ngrp  = wave & 3;            // 4 groups of 64 columns (HF=256)
    int mtile = (wave >> 2) & 2047;  // 2048 M-tiles
    int which = wave >> 13;          // 0=q, 1=k, 2=v

    const __bf16* Wt   = (which == 0) ? WtQ : (which == 1) ? WtK : WtV;
    const float*  bias = (which == 0) ? bq  : (which == 1) ? bk  : bv;

    int arow = mtile * 16 + lm;      // A row (per lane)

    v8f c[4] = {{}, {}, {}, {}};
    #pragma unroll
    for (int kc = 0; kc < 4; ++kc) {     // K = 128 = 4 * 32
        v16bf a = ld_frag_bf(xb + (size_t)arow * CC + kc * 32, hi);
        #pragma unroll
        for (int t = 0; t < 4; ++t) {
            int ncol = ngrp * 64 + t * 16 + lm;
            v16bf b = ld_frag_bf(Wt + (size_t)ncol * CC + kc * 32, hi);
            c[t] = __builtin_amdgcn_wmma_f32_16x16x32_bf16(false, a, false, b,
                                                           (short)0, c[t], false, false);
        }
    }
    #pragma unroll
    for (int t = 0; t < 4; ++t) {
        int ncol = ngrp * 64 + t * 16 + lm;
        float bvv = bias[ncol];
        if (which == 2) {                      // V: transposed [feature][pixel]
            #pragma unroll
            for (int r = 0; r < 8; ++r) {
                int drow = mtile * 16 + r + hi * 8;
                vbT[(size_t)ncol * NPIX + drow] = (__bf16)(c[t][r] + bvv);
            }
        } else {
            __bf16* ob = (which == 0) ? qb : kb;
            #pragma unroll
            for (int r = 0; r < 8; ++r) {
                int drow = mtile * 16 + r + hi * 8;
                ob[(size_t)drow * HF + ncol] = (__bf16)(c[t][r] + bvv);
            }
        }
    }
}

// ---------------------------------------------------------------------------
// Kernel 2: windowed attention. 1 block = 1 (batch, window); 8 waves = 8 heads
// scores+bias (WMMA) -> exp in LDS (unnormalized, bf16) -> P*V (WMMA)
// -> per-row 1/sum scaling in epilogue -> atomic scatter into acc
// ---------------------------------------------------------------------------
__global__ __launch_bounds__(256) void attn_kernel(
    const __bf16* __restrict__ qb,
    const __bf16* __restrict__ kb,
    const __bf16* __restrict__ vbT,
    const float* __restrict__ rpb,
    const float* __restrict__ Ey,
    const float* __restrict__ Ex,
    float* __restrict__ acc)
{
    __shared__ float  S_lds[HH * 64 * 64];     // 128 KB scores (f32)
    __shared__ __bf16 P_lds[HH * 64 * 64];     //  64 KB unnormalized probs
    __shared__ float  bias_lds[HH * 232];      // per-head rel-pos bias cache
    __shared__ float  isum_lds[HH * 64];       // per-row 1/sum

    const int head = threadIdx.x >> 5;
    const int lane = threadIdx.x & 31;
    const int hi = lane >> 4, lm = lane & 15;

    const int win = blockIdx.x;                // 0 .. B*NWIN*NWIN-1
    const int b   = win / (NWIN * NWIN);
    const int w   = win % (NWIN * NWIN);
    const int y0  = (w / NWIN) * 4;
    const int x0  = (w % NWIN) * 4;

    float*  S  = S_lds + head * 4096;
    __bf16* P  = P_lds + head * 4096;          // row stride 64
    float*  bC = bias_lds + head * 232;
    float*  iS = isum_lds + head * 64;

    // cache this head's 225 rel-pos bias values (wave-local)
    for (int i = lane; i < 225; i += 32) bC[i] = rpb[i * HH + head];

    auto tokPix = [&](int t) -> size_t {       // token -> flat pixel index
        return ((size_t)b * YY + (y0 + (t >> 3))) * XX + (x0 + (t & 7));
    };

    // ---- Q fragments (A operand), 4 M-tiles, K = d = 32: 2x b128 each ----
    v16bf qf[4];
    #pragma unroll
    for (int ti = 0; ti < 4; ++ti)
        qf[ti] = ld_frag_bf(qb + tokPix(ti * 16 + lm) * HF + head * DD, hi);

    // ---- scores S = Q*(K+pe)^T * rsd + bias ----
    const float rsd = 0.17677669529663687f;    // 1/sqrt(32)
    #pragma unroll
    for (int tj = 0; tj < 4; ++tj) {
        int m = tj * 16 + lm;                  // key token = B column (per lane)
        const __bf16* kr = kb + tokPix(m) * HF + head * DD;
        const float* pe = (head < 4)
            ? (Ey + (m >> 3) * 128 + head * DD)         // first HF/2 features
            : (Ex + (m & 7) * 128 + (head - 4) * DD);   // second HF/2 features
        // vectorized K load (2x b128) + packed pe add (4x b128 f32)
        v8bf klo = *(const v8bf*)(kr + (hi << 3));
        v8bf khi = *(const v8bf*)(kr + 16 + (hi << 3));
        float4 p0 = ((const float4*)(pe + (hi << 3)))[0];
        float4 p1 = ((const float4*)(pe + (hi << 3)))[1];
        float4 p2 = ((const float4*)(pe + 16 + (hi << 3)))[0];
        float4 p3 = ((const float4*)(pe + 16 + (hi << 3)))[1];
        v16bf bf;
        bf[0]  = (__bf16)((float)klo[0] + p0.x); bf[1]  = (__bf16)((float)klo[1] + p0.y);
        bf[2]  = (__bf16)((float)klo[2] + p0.z); bf[3]  = (__bf16)((float)klo[3] + p0.w);
        bf[4]  = (__bf16)((float)klo[4] + p1.x); bf[5]  = (__bf16)((float)klo[5] + p1.y);
        bf[6]  = (__bf16)((float)klo[6] + p1.z); bf[7]  = (__bf16)((float)klo[7] + p1.w);
        bf[8]  = (__bf16)((float)khi[0] + p2.x); bf[9]  = (__bf16)((float)khi[1] + p2.y);
        bf[10] = (__bf16)((float)khi[2] + p2.z); bf[11] = (__bf16)((float)khi[3] + p2.w);
        bf[12] = (__bf16)((float)khi[4] + p3.x); bf[13] = (__bf16)((float)khi[5] + p3.y);
        bf[14] = (__bf16)((float)khi[6] + p3.z); bf[15] = (__bf16)((float)khi[7] + p3.w);

        int cby = m >> 3, cbx = m & 7;
        #pragma unroll
        for (int ti = 0; ti < 4; ++ti) {
            v8f c = {};
            c = __builtin_amdgcn_wmma_f32_16x16x32_bf16(false, qf[ti], false, bf,
                                                        (short)0, c, false, false);
            #pragma unroll
            for (int r = 0; r < 8; ++r) {
                int n = ti * 16 + r + hi * 8;
                int dy = (n >> 3) - cby + 7, dx = (n & 7) - cbx + 7;
                S[n * 64 + m] = c[r] * rsd + bC[dy * 15 + dx];
            }
        }
    }
    __syncthreads();

    // ---- rowwise exp: each lane owns rows 2*lane, 2*lane+1 ----
    #pragma unroll
    for (int rr = 0; rr < 2; ++rr) {
        int n = lane * 2 + rr;
        const float4* Sr = (const float4*)(S + n * 64);
        float mx = -3.4e38f;
        #pragma unroll
        for (int q = 0; q < 16; ++q) {
            float4 t = Sr[q];
            mx = fmaxf(mx, fmaxf(fmaxf(t.x, t.y), fmaxf(t.z, t.w)));
        }
        float sum = 0.0f;
        v4bf* Pr = (v4bf*)(P + n * 64);
        #pragma unroll
        for (int q = 0; q < 16; ++q) {
            float4 t = Sr[q];
            float e0 = __expf(t.x - mx), e1 = __expf(t.y - mx);
            float e2 = __expf(t.z - mx), e3 = __expf(t.w - mx);
            sum += (e0 + e1) + (e2 + e3);
            v4bf pk;
            pk[0] = (__bf16)e0; pk[1] = (__bf16)e1;
            pk[2] = (__bf16)e2; pk[3] = (__bf16)e3;
            Pr[q] = pk;                         // b64 DS store
        }
        iS[n] = 1.0f / sum;
    }
    __syncthreads();

    // ---- O = P*V (K = 64 = 2x32, N = d = 32 = 2 tiles), scale rows by 1/sum --
    // V fragments from transposed V: each 8-token group = 8 consecutive pixels
    v16bf vf[2][2];                             // [kc][tj]
    #pragma unroll
    for (int kc = 0; kc < 2; ++kc)
        #pragma unroll
        for (int tj = 0; tj < 2; ++tj) {
            const __bf16* vr = vbT + (size_t)(head * DD + tj * 16 + lm) * NPIX;
            int t0 = kc * 32 + (hi << 3);       // token row group 0 (8 tokens)
            int t1 = t0 + 16;                   // token row group 1
            const __bf16* a0 = vr + tokPix(t0); // 8B-aligned (x0 multiple of 4)
            const __bf16* a1 = vr + tokPix(t1);
            v8bf lo = cat44(((const v4bf*)a0)[0], ((const v4bf*)a0)[1]);
            v8bf hh = cat44(((const v4bf*)a1)[0], ((const v4bf*)a1)[1]);
            vf[kc][tj] = __builtin_shufflevector(lo, hh, 0, 1, 2, 3, 4, 5, 6, 7,
                                                        8, 9, 10, 11, 12, 13, 14, 15);
        }

    #pragma unroll
    for (int ti = 0; ti < 4; ++ti) {
        v16bf pf0 = ld_frag_bf(P + (ti * 16 + lm) * 64, hi);        // LDS b128
        v16bf pf1 = ld_frag_bf(P + (ti * 16 + lm) * 64 + 32, hi);
        #pragma unroll
        for (int tj = 0; tj < 2; ++tj) {
            v8f c = {};
            c = __builtin_amdgcn_wmma_f32_16x16x32_bf16(false, pf0, false, vf[0][tj],
                                                        (short)0, c, false, false);
            c = __builtin_amdgcn_wmma_f32_16x16x32_bf16(false, pf1, false, vf[1][tj],
                                                        (short)0, c, false, false);
            #pragma unroll
            for (int r = 0; r < 8; ++r) {
                int tok = ti * 16 + r + hi * 8;
                float o = c[r] * iS[tok];
                atomicAdd(&acc[tokPix(tok) * HF + head * DD + tj * 16 + lm], o);
            }
        }
    }
}

// ---------------------------------------------------------------------------
// Kernel 3: output projection (M=32768, K=256, N=128) + bias + skip
// register-blocked 16M x 64N per wave; overlap-mean folded analytically.
// ---------------------------------------------------------------------------
__global__ __launch_bounds__(256) void outproj_kernel(
    const float* __restrict__ acc,
    const __bf16* __restrict__ WoT, const float* __restrict__ bo,
    const float* __restrict__ x, float* __restrict__ out)
{
    int wave = (int)((blockIdx.x * blockDim.x + threadIdx.x) >> 5);
    int lane = threadIdx.x & 31;
    int hi = lane >> 4, lm = lane & 15;

    int ngrp  = wave & 1;            // 2 groups of 64 columns (C=128)
    int mtile = wave >> 1;           // 2048 M-tiles

    int arow = mtile * 16 + lm;
    int y = (arow >> 7) & 127, xq = arow & 127;
    float cy = (y  < 4 || y  >= 124) ? 1.0f : 2.0f;
    float cx = (xq < 4 || xq >= 124) ? 1.0f : 2.0f;
    float ascale = 1.0f / (cy * cx);

    v8f c[4] = {{}, {}, {}, {}};
    #pragma unroll
    for (int kc = 0; kc < 8; ++kc) {      // K = 256 = 8 * 32
        v16bf a = ld_frag_f32(acc + (size_t)arow * HF + kc * 32, hi, ascale);
        #pragma unroll
        for (int t = 0; t < 4; ++t) {
            int ncol = ngrp * 64 + t * 16 + lm;
            v16bf b = ld_frag_bf(WoT + (size_t)ncol * HF + kc * 32, hi);
            c[t] = __builtin_amdgcn_wmma_f32_16x16x32_bf16(false, a, false, b,
                                                           (short)0, c[t], false, false);
        }
    }
    #pragma unroll
    for (int t = 0; t < 4; ++t) {
        int ncol = ngrp * 64 + t * 16 + lm;
        float bias = bo[ncol];
        #pragma unroll
        for (int r = 0; r < 8; ++r) {
            int drow = mtile * 16 + r + hi * 8;
            out[(size_t)drow * CC + ncol] = c[t][r] + bias + x[(size_t)drow * CC + ncol];
        }
    }
}

// ---------------------------------------------------------------------------
// Launch
// ---------------------------------------------------------------------------
extern "C" void kernel_launch(void* const* d_in, const int* in_sizes, int n_in,
                              void* d_out, int out_size, void* d_ws, size_t ws_size,
                              hipStream_t stream) {
    (void)in_sizes; (void)n_in; (void)out_size; (void)ws_size;

    const float* x   = (const float*)d_in[0];
    const float* Wq  = (const float*)d_in[1];
    const float* bq  = (const float*)d_in[2];
    const float* Wk  = (const float*)d_in[3];
    const float* bk  = (const float*)d_in[4];
    const float* Wv  = (const float*)d_in[5];
    const float* bv  = (const float*)d_in[6];
    const float* Wo  = (const float*)d_in[7];
    const float* bo  = (const float*)d_in[8];
    const float* rpb = (const float*)d_in[9];
    const float* Ey  = (const float*)d_in[10];
    const float* Ex  = (const float*)d_in[11];
    float* out = (float*)d_out;

    // workspace layout (bytes)
    const size_t nQKV = (size_t)NPIX * HF;     // 8,388,608 elements
    const size_t nX   = (size_t)NPIX * CC;     // 4,194,304 elements
    char* ws = (char*)d_ws;
    __bf16* xb  = (__bf16*)(ws);                               //  8 MB
    __bf16* qb  = (__bf16*)(ws + nX * 2);                      // 16 MB
    __bf16* kb  = (__bf16*)(ws + nX * 2 + nQKV * 2);           // 16 MB
    __bf16* vbT = (__bf16*)(ws + nX * 2 + nQKV * 4);           // 16 MB (transposed)
    float*  ac  = (float*) (ws + nX * 2 + nQKV * 6);           // 32 MB
    __bf16* WtQ = (__bf16*)(ws + nX * 2 + nQKV * 6 + nQKV * 4);        // 64 KB
    __bf16* WtK = WtQ + (size_t)CC * HF;
    __bf16* WtV = WtK + (size_t)CC * HF;
    __bf16* WoT = WtV + (size_t)CC * HF;                       // 128x256 -> 64 KB

    // 0) prep: zero accumulator, convert x, transpose+convert weights
    zero_kernel<<<1024, 256, 0, stream>>>(ac, nQKV);
    cvt4_kernel<<<(int)((nX / 4 + 255) / 256), 256, 0, stream>>>(x, xb, nX / 4);
    transpose_kernel<<<(CC * HF + 255) / 256, 256, 0, stream>>>(Wq, WtQ, CC, HF);
    transpose_kernel<<<(CC * HF + 255) / 256, 256, 0, stream>>>(Wk, WtK, CC, HF);
    transpose_kernel<<<(CC * HF + 255) / 256, 256, 0, stream>>>(Wv, WtV, CC, HF);
    transpose_kernel<<<(HF * CC + 255) / 256, 256, 0, stream>>>(Wo, WoT, HF, CC);

    // 1) QKV projections: 3 * 2048 * 4 wave-tiles (16Mx64N each) / 8 per block
    qkv_kernel<<<(3 * 2048 * 4) / 8, 256, 0, stream>>>(
        xb, WtQ, bq, WtK, bk, WtV, bv, qb, kb, vbT);

    // 2) attention: one block per (batch, window)
    attn_kernel<<<BB * NWIN * NWIN, 256, 0, stream>>>(
        qb, kb, vbT, rpb, Ey, Ex, ac);

    // 3) output projection + skip: 2048 * 2 wave-tiles / 8 per block
    outproj_kernel<<<(2048 * 2) / 8, 256, 0, stream>>>(ac, WoT, bo, x, out);
}
